// DRAW_30150670417921
// MI455X (gfx1250) — matrix operand order repaired and checked by reading, not compiled, not run on hardware
//
#include <hip/hip_runtime.h>
#include <hip/hip_bf16.h>

// ---------------- problem constants ----------------
#define T_STEPS 16
#define A_DIM   64
#define B_DIM   64
#define N_ATT   12
#define REP_DIM 100
#define ENC_DIM 800
#define DEC_DIM 800
#define BATCH   512
#define IMG     (A_DIM * B_DIM)        // 4096
#define ENC_IN  (2 * N_ATT * N_ATT + DEC_DIM)  // 1088
#define FILT    (N_ATT * A_DIM)        // 768

typedef __attribute__((ext_vector_type(16))) _Float16 v16h;
typedef __attribute__((ext_vector_type(8)))  _Float16 v8h;
typedef __attribute__((ext_vector_type(8)))  float    v8f;

#define FLAG_BIAS 1
#define FLAG_ACC  2

// ---------------- f32 -> f16 cast ----------------
__global__ void draw_cast_f16(const float* __restrict__ src,
                              _Float16* __restrict__ dst, int n) {
    int i = blockIdx.x * blockDim.x + threadIdx.x;
    if (i < n) dst[i] = (_Float16)src[i];
}

// ---------------- LDS tile stage (f16, 64-wide K tiles) ----------------
__device__ inline void draw_stage_tile64(_Float16* __restrict__ dst,
                                         const _Float16* __restrict__ src,
                                         int ld, int rows, int K,
                                         int row0, int k0, int tileRows, int tid) {
    const int nch = (tileRows * 64) / 8;   // 8 f16 per chunk
    for (int ch = tid; ch < nch; ch += 256) {
        int r  = ch >> 3;                  // 8 chunks per 64-wide row
        int kc = (ch & 7) << 3;
        _Float16* d = dst + r * 64 + kc;
        int gr = row0 + r;
        int gk = k0 + kc;
        if (gr < rows) {
            size_t off = (size_t)gr * ld + gk;
            if (gk + 8 <= K && (off & 7) == 0) {
                *(v8h*)d = *(const v8h*)(src + off);
            } else {
                for (int e = 0; e < 8; ++e)
                    d[e] = (gk + e < K) ? src[off + e] : (_Float16)0.f;
            }
        } else {
            v8h z = {};
            *(v8h*)d = z;
        }
    }
}

// ---------------- WMMA GEMM: OUT[M,N] = A[M,K] @ W[N,K]^T (+bias / +=OUT) ----
// 256 threads = 8 waves; block tile 128x64; K tile 64 (2 WMMA sub-steps).
__global__ __launch_bounds__(256)
void draw_gemm_wmma(float* __restrict__ OUT,
                    const _Float16* __restrict__ Amat,
                    const _Float16* __restrict__ Wmat,
                    const float* __restrict__ bias,
                    int M, int Nout, int K, int flags) {
    __shared__ __align__(16) _Float16 As[128 * 64];   // 16 KB
    __shared__ __align__(16) _Float16 Bs[64 * 64];    //  8 KB

    const int tid  = threadIdx.x;
    const int lane = tid & 31;
    const int wave = tid >> 5;          // 0..7
    const int half = lane >> 4;         // 0 or 1
    const int lm   = lane & 15;
    const int row0 = blockIdx.y * 128;
    const int col0 = blockIdx.x * 64;

    v8f acc[4] = {};

    for (int k0 = 0; k0 < K; k0 += 64) {
        draw_stage_tile64(As, Amat, K, M,    K, row0, k0, 128, tid);
        draw_stage_tile64(Bs, Wmat, K, Nout, K, col0, k0, 64,  tid);
        if (k0 + 64 < K) {
            int pa = row0 + (tid >> 1);            // 0..127
            int pw = col0 + (tid & 63);
            if (pw >= Nout) pw = Nout - 1;
            __builtin_prefetch(Amat + (size_t)pa * K + k0 + 64, 0, 1);
            __builtin_prefetch(Wmat + (size_t)pw * K + k0 + 64, 0, 1);
        }
        __syncthreads();

        // Two 32-deep WMMA sub-steps out of the staged 64-wide tile.
        const _Float16* ar = As + (wave * 16 + lm) * 64;
#pragma unroll
        for (int sub = 0; sub < 2; ++sub) {
            const int ks = sub * 32;
            // A fragment (ISA 16-bit A 16x32 layout): row = wave*16+lm,
            // K chunks {half*8..+7} and {16+half*8..+7} within the sub-tile.
            v8h alo = *(const v8h*)(ar + ks + half * 8);
            v8h ahi = *(const v8h*)(ar + ks + 16 + half * 8);
            v16h afrag = __builtin_shufflevector(alo, ahi,
                0, 1, 2, 3, 4, 5, 6, 7, 8, 9, 10, 11, 12, 13, 14, 15);

            const int kb = ks + half * 8;    // B: same K chunks as A: {half*8..+7} and {16+half*8..+7}
#pragma unroll
            for (int i = 0; i < 4; ++i) {
                const _Float16* br = Bs + (i * 16 + lm) * 64 + kb;
                v8h blo = *(const v8h*)br;
                v8h bhi = *(const v8h*)(br + 16);
                v16h bfrag = __builtin_shufflevector(blo, bhi,
                    0, 1, 2, 3, 4, 5, 6, 7, 8, 9, 10, 11, 12, 13, 14, 15);
                acc[i] = __builtin_amdgcn_wmma_f32_16x16x32_f16(
                    false, afrag, false, bfrag, (short)0, acc[i], false, false);
            }
        }
        __syncthreads();
    }

    // C/D layout: lane -> col=lm, VGPR r -> row = half*8 + r
#pragma unroll
    for (int i = 0; i < 4; ++i) {
        int col = col0 + i * 16 + lm;
        if (col >= Nout) continue;
        float badd = (flags & FLAG_BIAS) ? bias[col] : 0.f;
#pragma unroll
        for (int r = 0; r < 8; ++r) {
            int row = row0 + wave * 16 + half * 8 + r;
            if (row < M) {
                size_t o = (size_t)row * Nout + col;
                float v = acc[i][r] + badd;
                if (flags & FLAG_ACC) v += OUT[o];
                OUT[o] = v;
            }
        }
    }
}

// ---------------- attention filter: h_dec -> Fx, Fy, gamma ----------------
__global__ __launch_bounds__(256)
void draw_filter(const float* __restrict__ h, const float* __restrict__ rW,
                 const float* __restrict__ rb,
                 float* __restrict__ Fx, float* __restrict__ Fy,
                 float* __restrict__ gamma_out) {
    __shared__ float rbuf[5 * 256];
    __shared__ float par[5];
    __shared__ float sFx[FILT], sFy[FILT];
    __shared__ float sinv[2 * N_ATT];
    const int b = blockIdx.x, tid = threadIdx.x;

    float a0 = 0, a1 = 0, a2 = 0, a3 = 0, a4 = 0;
    for (int k = tid; k < DEC_DIM; k += 256) {
        float hv = h[(size_t)b * DEC_DIM + k];
        a0 += hv * rW[0 * DEC_DIM + k];
        a1 += hv * rW[1 * DEC_DIM + k];
        a2 += hv * rW[2 * DEC_DIM + k];
        a3 += hv * rW[3 * DEC_DIM + k];
        a4 += hv * rW[4 * DEC_DIM + k];
    }
    rbuf[0 * 256 + tid] = a0; rbuf[1 * 256 + tid] = a1; rbuf[2 * 256 + tid] = a2;
    rbuf[3 * 256 + tid] = a3; rbuf[4 * 256 + tid] = a4;
    __syncthreads();
    for (int s = 128; s > 0; s >>= 1) {
        if (tid < s)
            for (int j = 0; j < 5; ++j) rbuf[j * 256 + tid] += rbuf[j * 256 + tid + s];
        __syncthreads();
    }
    if (tid == 0) {
        float gx    = rbuf[0]        + rb[0];
        float gy    = rbuf[1 * 256]  + rb[1];
        float var   = __expf(rbuf[2 * 256] + rb[2]);
        float delta = ((float)(A_DIM - 1) / (float)(N_ATT - 1)) * __expf(rbuf[3 * 256] + rb[3]);
        float gam   = __expf(rbuf[4 * 256] + rb[4]);
        par[0] = 0.5f * (A_DIM + 1) * (gx + 1.f);
        par[1] = 0.5f * (B_DIM + 1) * (gy + 1.f);
        par[2] = 1.f / (2.f * var);
        par[3] = delta;
        gamma_out[b] = gam;
    }
    __syncthreads();
    for (int idx = tid; idx < FILT; idx += 256) {
        int n = idx >> 6, a = idx & 63;
        float off = ((float)n - (float)N_ATT * 0.5f - 0.5f) * par[3];
        float dx = (float)a - (par[0] + off);
        float dy = (float)a - (par[1] + off);
        sFx[idx] = __expf(-dx * dx * par[2]);
        sFy[idx] = __expf(-dy * dy * par[2]);
    }
    __syncthreads();
    if (tid < N_ATT) {
        float s = 0;
        for (int a = 0; a < A_DIM; ++a) s += sFx[tid * 64 + a];
        sinv[tid] = 1.f / (s + 1e-9f);
    } else if (tid >= 32 && tid < 32 + N_ATT) {
        int n = tid - 32;
        float s = 0;
        for (int a = 0; a < A_DIM; ++a) s += sFy[n * 64 + a];
        sinv[N_ATT + n] = 1.f / (s + 1e-9f);
    }
    __syncthreads();
    for (int idx = tid; idx < FILT; idx += 256) {
        int n = idx >> 6;
        Fx[(size_t)b * FILT + idx] = sFx[idx] * sinv[n];
        Fy[(size_t)b * FILT + idx] = sFy[idx] * sinv[N_ATT + n];
    }
}

// ---------------- read attention + pack encoder input (f16) ----------------
__global__ __launch_bounds__(256)
void draw_read(const float* __restrict__ x, const float* __restrict__ pre_c,
               const float* __restrict__ Fx, const float* __restrict__ Fy,
               const float* __restrict__ gamma, const float* __restrict__ h_dec,
               _Float16* __restrict__ enc_in) {
    __shared__ float sFx[FILT], sFy[FILT];
    __shared__ float sImg[IMG];
    __shared__ float sT[FILT];
    const int b = blockIdx.x, tid = threadIdx.x;
    for (int i = tid; i < FILT; i += 256) {
        sFx[i] = Fx[(size_t)b * FILT + i];
        sFy[i] = Fy[(size_t)b * FILT + i];
    }
    const float gam = gamma[b];
    for (int pass = 0; pass < 2; ++pass) {
        __syncthreads();
        for (int i = tid; i < IMG; i += 256) {
            float v = x[(size_t)b * IMG + i];
            if (pass) {
                float pc = pre_c[(size_t)b * IMG + i];
                v -= 1.f / (1.f + __expf(-pc));       // x_hat = x - sigmoid(pre_c)
            }
            sImg[i] = v;
        }
        __syncthreads();
        for (int i = tid; i < FILT; i += 256) {       // t = Fy @ img
            int n = i >> 6, a = i & 63;
            float s = 0;
            for (int bb = 0; bb < B_DIM; ++bb) s += sFy[n * 64 + bb] * sImg[bb * 64 + a];
            sT[i] = s;
        }
        __syncthreads();
        if (tid < N_ATT * N_ATT) {                    // g = t @ Fx^T
            int n = tid / N_ATT, m = tid % N_ATT;
            float s = 0;
            for (int a = 0; a < A_DIM; ++a) s += sT[n * 64 + a] * sFx[m * 64 + a];
            enc_in[(size_t)b * ENC_IN + pass * (N_ATT * N_ATT) + tid] = (_Float16)(s * gam);
        }
    }
    __syncthreads();
    for (int j = tid; j < DEC_DIM; j += 256)          // pack h_dec
        enc_in[(size_t)b * ENC_IN + 2 * N_ATT * N_ATT + j] =
            (_Float16)h_dec[(size_t)b * DEC_DIM + j];
}

// ---------------- LSTM cell elementwise ----------------
__global__ void draw_lstm(const float* __restrict__ gates, float* __restrict__ h,
                          float* __restrict__ c, _Float16* __restrict__ h_half,
                          int H, int n) {
    int idx = blockIdx.x * blockDim.x + threadIdx.x;
    if (idx >= n) return;
    int b = idx / H, j = idx - b * H;
    const float* g = gates + (size_t)b * 4 * H;
    float gi = g[j], gf = g[H + j], gg = g[2 * H + j], go = g[3 * H + j];
    float si = 1.f / (1.f + __expf(-gi));
    float sf = 1.f / (1.f + __expf(-gf));
    float so = 1.f / (1.f + __expf(-go));
    float c2 = sf * c[idx] + si * tanhf(gg);
    float h2 = so * tanhf(c2);
    c[idx] = c2; h[idx] = h2; h_half[idx] = (_Float16)h2;
}

// ---------------- z = mu + n_t * exp(logsig) -> f16 ----------------
__global__ void draw_z(const float* __restrict__ mu, const float* __restrict__ ls,
                       const float* __restrict__ noise_t, _Float16* __restrict__ z_h,
                       int n) {
    int i = blockIdx.x * blockDim.x + threadIdx.x;
    if (i < n) z_h[i] = (_Float16)(mu[i] + noise_t[i] * __expf(ls[i]));
}

// ---------------- write attention + canvas update ----------------
__global__ __launch_bounds__(256)
void draw_write(const float* __restrict__ wt, const float* __restrict__ Fx,
                const float* __restrict__ Fy, const float* __restrict__ gamma,
                float* __restrict__ pre_c, float* __restrict__ out_t) {
    __shared__ float sFx[FILT], sFy[FILT], sW[N_ATT * N_ATT], sT[FILT];
    const int b = blockIdx.x, tid = threadIdx.x;
    for (int i = tid; i < FILT; i += 256) {
        sFx[i] = Fx[(size_t)b * FILT + i];
        sFy[i] = Fy[(size_t)b * FILT + i];
    }
    if (tid < N_ATT * N_ATT) sW[tid] = wt[(size_t)b * (N_ATT * N_ATT) + tid];
    float invg = 1.f / gamma[b];
    __syncthreads();
    for (int i = tid; i < FILT; i += 256) {           // t = w @ Fx
        int n = i >> 6, a = i & 63;
        float s = 0;
        for (int m = 0; m < N_ATT; ++m) s += sW[n * N_ATT + m] * sFx[m * 64 + a];
        sT[i] = s;
    }
    __syncthreads();
    for (int i = tid; i < IMG; i += 256) {            // wimg = Fy^T @ t
        int bb = i >> 6, a = i & 63;
        float s = 0;
        for (int n = 0; n < N_ATT; ++n) s += sFy[n * 64 + bb] * sT[n * 64 + a];
        float cv = pre_c[(size_t)b * IMG + i] + s * invg;
        pre_c[(size_t)b * IMG + i] = cv;
        out_t[(size_t)b * IMG + i] = cv;
    }
}

// ---------------- host-side orchestration ----------------
extern "C" void kernel_launch(void* const* d_in, const int* in_sizes, int n_in,
                              void* d_out, int out_size, void* d_ws, size_t ws_size,
                              hipStream_t stream) {
    const float* x        = (const float*)d_in[0];
    const float* noise    = (const float*)d_in[1];
    const float* enc_Wih  = (const float*)d_in[2];
    const float* enc_Whh  = (const float*)d_in[3];
    const float* enc_b    = (const float*)d_in[4];
    const float* dec_Wih  = (const float*)d_in[5];
    const float* dec_Whh  = (const float*)d_in[6];
    const float* dec_b    = (const float*)d_in[7];
    const float* mu_W     = (const float*)d_in[8];
    const float* mu_b     = (const float*)d_in[9];
    const float* sig_W    = (const float*)d_in[10];
    const float* sig_b    = (const float*)d_in[11];
    const float* read_W   = (const float*)d_in[12];
    const float* read_b   = (const float*)d_in[13];
    const float* write_W  = (const float*)d_in[14];
    const float* write_b  = (const float*)d_in[15];
    float* out = (float*)d_out;

    char* ws = (char*)d_ws;
    size_t off = 0;
    auto alloc = [&](size_t bytes) -> void* {
        void* p = ws + off;
        off = (off + bytes + 255) & ~(size_t)255;
        return p;
    };

    // f16 weights
    _Float16* encWih_h = (_Float16*)alloc((size_t)4 * ENC_DIM * ENC_IN * 2);
    _Float16* encWhh_h = (_Float16*)alloc((size_t)4 * ENC_DIM * ENC_DIM * 2);
    _Float16* decWih_h = (_Float16*)alloc((size_t)4 * DEC_DIM * REP_DIM * 2);
    _Float16* decWhh_h = (_Float16*)alloc((size_t)4 * DEC_DIM * DEC_DIM * 2);
    _Float16* muW_h    = (_Float16*)alloc((size_t)REP_DIM * ENC_DIM * 2);
    _Float16* sigW_h   = (_Float16*)alloc((size_t)REP_DIM * ENC_DIM * 2);
    _Float16* writeW_h = (_Float16*)alloc((size_t)N_ATT * N_ATT * DEC_DIM * 2);
    // state / activations
    float* pre_c   = (float*)alloc((size_t)BATCH * IMG * 4);
    float* h_enc   = (float*)alloc((size_t)BATCH * ENC_DIM * 4);
    float* c_enc   = (float*)alloc((size_t)BATCH * ENC_DIM * 4);
    float* h_dec   = (float*)alloc((size_t)BATCH * DEC_DIM * 4);
    float* c_dec   = (float*)alloc((size_t)BATCH * DEC_DIM * 4);
    _Float16* h_enc_h = (_Float16*)alloc((size_t)BATCH * ENC_DIM * 2);
    _Float16* h_dec_h = (_Float16*)alloc((size_t)BATCH * DEC_DIM * 2);
    _Float16* enc_in_h = (_Float16*)alloc((size_t)BATCH * ENC_IN * 2);
    _Float16* z_h   = (_Float16*)alloc((size_t)BATCH * REP_DIM * 2);
    float* gates   = (float*)alloc((size_t)BATCH * 4 * ENC_DIM * 4);
    float* mu_buf  = (float*)alloc((size_t)BATCH * REP_DIM * 4);
    float* sig_buf = (float*)alloc((size_t)BATCH * REP_DIM * 4);
    float* Fx      = (float*)alloc((size_t)BATCH * FILT * 4);
    float* Fy      = (float*)alloc((size_t)BATCH * FILT * 4);
    float* gamma   = (float*)alloc((size_t)BATCH * 4);
    float* wt_buf  = (float*)alloc((size_t)BATCH * N_ATT * N_ATT * 4);
    (void)ws_size; (void)n_in; (void)in_sizes; (void)out_size;

    auto cast = [&](const float* s, _Float16* d, int n) {
        draw_cast_f16<<<(n + 255) / 256, 256, 0, stream>>>(s, d, n);
    };
    cast(enc_Wih, encWih_h, 4 * ENC_DIM * ENC_IN);
    cast(enc_Whh, encWhh_h, 4 * ENC_DIM * ENC_DIM);
    cast(dec_Wih, decWih_h, 4 * DEC_DIM * REP_DIM);
    cast(dec_Whh, decWhh_h, 4 * DEC_DIM * DEC_DIM);
    cast(mu_W,    muW_h,    REP_DIM * ENC_DIM);
    cast(sig_W,   sigW_h,   REP_DIM * ENC_DIM);
    cast(write_W, writeW_h, N_ATT * N_ATT * DEC_DIM);

    hipMemsetAsync(pre_c,   0, (size_t)BATCH * IMG * 4,     stream);
    hipMemsetAsync(h_enc,   0, (size_t)BATCH * ENC_DIM * 4, stream);
    hipMemsetAsync(c_enc,   0, (size_t)BATCH * ENC_DIM * 4, stream);
    hipMemsetAsync(h_dec,   0, (size_t)BATCH * DEC_DIM * 4, stream);
    hipMemsetAsync(c_dec,   0, (size_t)BATCH * DEC_DIM * 4, stream);
    hipMemsetAsync(h_enc_h, 0, (size_t)BATCH * ENC_DIM * 2, stream);
    hipMemsetAsync(h_dec_h, 0, (size_t)BATCH * DEC_DIM * 2, stream);

    auto gemm = [&](float* C, const _Float16* Ah, const _Float16* Wh,
                    const float* bias, int M, int N, int K, int flags) {
        dim3 g((N + 63) / 64, (M + 127) / 128);
        draw_gemm_wmma<<<g, 256, 0, stream>>>(C, Ah, Wh, bias, M, N, K, flags);
    };

    const int nSt = BATCH * DEC_DIM;
    for (int t = 0; t < T_STEPS; ++t) {
        // read filter from current h_dec
        draw_filter<<<BATCH, 256, 0, stream>>>(h_dec, read_W, read_b, Fx, Fy, gamma);
        draw_read<<<BATCH, 256, 0, stream>>>(x, pre_c, Fx, Fy, gamma, h_dec, enc_in_h);
        // encoder gates
        gemm(gates, enc_in_h, encWih_h, enc_b, BATCH, 4 * ENC_DIM, ENC_IN, FLAG_BIAS);
        gemm(gates, h_enc_h,  encWhh_h, nullptr, BATCH, 4 * ENC_DIM, ENC_DIM, FLAG_ACC);
        draw_lstm<<<(nSt + 255) / 256, 256, 0, stream>>>(gates, h_enc, c_enc, h_enc_h,
                                                         ENC_DIM, nSt);
        // latent
        gemm(mu_buf,  h_enc_h, muW_h,  mu_b,  BATCH, REP_DIM, ENC_DIM, FLAG_BIAS);
        gemm(sig_buf, h_enc_h, sigW_h, sig_b, BATCH, REP_DIM, ENC_DIM, FLAG_BIAS);
        draw_z<<<(BATCH * REP_DIM + 255) / 256, 256, 0, stream>>>(
            mu_buf, sig_buf, noise + (size_t)t * BATCH * REP_DIM, z_h, BATCH * REP_DIM);
        // decoder gates
        gemm(gates, z_h,     decWih_h, dec_b,  BATCH, 4 * DEC_DIM, REP_DIM, FLAG_BIAS);
        gemm(gates, h_dec_h, decWhh_h, nullptr, BATCH, 4 * DEC_DIM, DEC_DIM, FLAG_ACC);
        draw_lstm<<<(nSt + 255) / 256, 256, 0, stream>>>(gates, h_dec, c_dec, h_dec_h,
                                                         DEC_DIM, nSt);
        // write patch + filter from updated h_dec
        gemm(wt_buf, h_dec_h, writeW_h, write_b, BATCH, N_ATT * N_ATT, DEC_DIM, FLAG_BIAS);
        draw_filter<<<BATCH, 256, 0, stream>>>(h_dec, read_W, read_b, Fx, Fy, gamma);
        draw_write<<<BATCH, 256, 0, stream>>>(wt_buf, Fx, Fy, gamma, pre_c,
                                              out + (size_t)t * BATCH * IMG);
    }
}
